// SingleHeadAttention_764504178963
// MI455X (gfx1250) — matrix operand order repaired and checked by reading, not compiled, not run on hardware
//
#include <hip/hip_runtime.h>
#include <hip/hip_bf16.h>

#define B_ 4
#define T_ 2048
#define C_ 1024
#define H_ 1024

typedef __attribute__((ext_vector_type(16))) __bf16 v16bf;
typedef __attribute__((ext_vector_type(8)))  float  v8f;
typedef __attribute__((ext_vector_type(4)))  float  f32x4;
typedef __attribute__((ext_vector_type(4)))  unsigned int u32x4;
typedef __attribute__((ext_vector_type(4)))  int    i32x4;
typedef __attribute__((ext_vector_type(8)))  int    i32x8;

union V16U { v16bf v; f32x4 f4[2]; };

__device__ __forceinline__ v8f wmma_bf16(v16bf a, v16bf b, v8f c) {
  return __builtin_amdgcn_wmma_f32_16x16x32_bf16(false, a, false, b, (short)0, c,
                                                 false, false);
}

// ---------------------------------------------------------------------------
// TDM: 1-D tile load (ndw 4-byte elements, contiguous) global -> LDS.
// D# group0: count=1, lds_addr, 57-bit global addr, type=2 ("image").
// D# group1: data_size=4B, tensor_dim0=ndw, tensor_dim1=1, tile_dim0=ndw,
//            tensor_dim0_stride=ndw.  Groups 2/3 unused (<=2D tensor).
// ---------------------------------------------------------------------------
__device__ __forceinline__ void tdm_load_1d(unsigned lds_off, const void* gptr,
                                            unsigned ndw) {
  unsigned long long ga = (unsigned long long)(uintptr_t)gptr;
  u32x4 g0;
  g0[0] = 1u;                                             // count=1, user mode
  g0[1] = lds_off;                                        // lds_addr (bytes)
  g0[2] = (unsigned)ga;                                   // global_addr[31:0]
  g0[3] = (unsigned)((ga >> 32) & 0x1FFFFFFull) | (2u << 30);  // [56:32]|type=2
  i32x8 g1 = (i32x8)0;
  g1[0] = (2 << 16);                                      // data_size = 4B
  g1[1] = (int)((ndw & 0xFFFFu) << 16);                   // tensor_dim0[15:0]
  g1[2] = (int)(((ndw >> 16) & 0xFFFFu) | (1u << 16));    // dim0 hi | dim1=1
  g1[3] = (int)((ndw & 0xFFFFu) << 16);                   // tile_dim0
  g1[5] = (int)ndw;                                       // tensor_dim0_stride
  i32x4 gz = (i32x4)0;
#if __clang_major__ >= 23
  __builtin_amdgcn_tensor_load_to_lds(g0, g1, gz, gz, (i32x8)0, 0);
#else
  __builtin_amdgcn_tensor_load_to_lds(g0, g1, gz, gz, 0);
#endif
}

// ---------------------------------------------------------------------------
// Kernel 1: K = x*Wk^T (row-major bf16), VT = (x*Wv^T)^T (bf16 [B,H,T]).
// Each wave: one 16x64 slab (4 N-tiles) -> A fragment reused 4x.
// ---------------------------------------------------------------------------
__global__ void __launch_bounds__(128)
proj_kv_kernel(const float* __restrict__ x, const float* __restrict__ Wk,
               const float* __restrict__ Wv, __bf16* __restrict__ Kb,
               __bf16* __restrict__ VTb) {
  const int lane = threadIdx.x & 31;
  const int hi   = lane >> 4;
  const int wid  = blockIdx.x * (blockDim.x >> 5) + (threadIdx.x >> 5);

  const int ng  = wid & 15;          // group of 4 N tiles (64 cols of H)
  const int mt  = (wid >> 4) & 511;  // M tile over B*T
  const int mat = wid >> 13;         // 0 = K, 1 = V
  const float* __restrict__ W = mat ? Wv : Wk;

  const int mbase = mt * 16;
  const int nbase = ng * 64;
  const int arow  = mbase + (lane & 15);
  const int a_c0  = hi ? 8 : 0;
  const int b_c0  = hi ? 8 : 0;

  v8f acc[4] = {{}, {}, {}, {}};
  for (int c = 0; c < C_; c += 32) {
    v16bf av;
    {
      const f32x4* p0 = (const f32x4*)(x + (size_t)arow * C_ + c + a_c0);
      const f32x4* p1 = (const f32x4*)(x + (size_t)arow * C_ + c + a_c0 + 16);
      f32x4 a0 = p0[0], a1 = p0[1], a2 = p1[0], a3 = p1[1];
#pragma unroll
      for (int i = 0; i < 4; ++i) {
        av[i]      = (__bf16)a0[i];
        av[4 + i]  = (__bf16)a1[i];
        av[8 + i]  = (__bf16)a2[i];
        av[12 + i] = (__bf16)a3[i];
      }
    }
#pragma unroll
    for (int n = 0; n < 4; ++n) {
      const int brow = nbase + n * 16 + (lane & 15);
      const f32x4* p = (const f32x4*)(W + (size_t)brow * C_ + c + b_c0);
      f32x4 b0 = p[0], b1 = p[1], b2 = p[4], b3 = p[5];
      v16bf bv;
#pragma unroll
      for (int i = 0; i < 4; ++i) {
        bv[i]      = (__bf16)b0[i];
        bv[4 + i]  = (__bf16)b1[i];
        bv[8 + i]  = (__bf16)b2[i];
        bv[12 + i] = (__bf16)b3[i];
      }
      acc[n] = wmma_bf16(av, bv, acc[n]);
    }
  }

  if (mat == 0) {
#pragma unroll
    for (int n = 0; n < 4; ++n)
#pragma unroll
      for (int j = 0; j < 8; ++j) {
        int row = mbase + j + (hi << 3);
        Kb[(size_t)row * H_ + nbase + n * 16 + (lane & 15)] = (__bf16)acc[n][j];
      }
  } else {
    int b  = mbase / T_;
    int t0 = (mbase % T_) + (hi << 3);
#pragma unroll
    for (int n = 0; n < 4; ++n) {
      int h = nbase + n * 16 + (lane & 15);
      union { __bf16 h8[8]; f32x4 f4; } u;
#pragma unroll
      for (int j = 0; j < 8; ++j) u.h8[j] = (__bf16)acc[n][j];
      *(f32x4*)(VTb + ((size_t)b * H_ + h) * T_ + t0) = u.f4;
    }
  }
}

// ---------------------------------------------------------------------------
// Kernel 2: S = scale * Q K^T (causal), bf16 scores + per-row (m, l) stats.
// Per-wave Q tile staged via TDM (tensor_load_to_lds) into its own 32KB LDS
// slice; two key tiles per A fragment to halve LDS read traffic.
// ---------------------------------------------------------------------------
__global__ void __launch_bounds__(128)
scores_kernel(const __bf16* __restrict__ Kb, __bf16* __restrict__ Sb,
              float* __restrict__ mbuf, float* __restrict__ lbuf) {
  extern __shared__ char smem[];
  const int lane = threadIdx.x & 31;
  const int hi   = lane >> 4;
  const int w    = threadIdx.x >> 5;
  const int b    = blockIdx.x >> 5;  // 32 blocks per batch
  const int qg   = blockIdx.x & 31;  // 64-row query group

  const int qtile = qg * 4 + w;
  const int qbase = qtile * 16;
  const int qslot = w * 16;

  // Stage this wave's 16 Q rows (32KB) LDS slice with the Tensor Data Mover.
  {
    unsigned lds_off = (unsigned)(uintptr_t)smem + (unsigned)(w * 32768);
    const void* gsrc = Kb + ((size_t)b * T_ + qbase) * H_;
    tdm_load_1d(lds_off, gsrc, 32768 / 4);
    __builtin_amdgcn_s_wait_tensorcnt(0);
  }

  const int a_k0 = hi ? 8 : 0;
  const int b_k0 = hi ? 8 : 0;
  const int am   = lane & 15;

  float m[8], l[8];
#pragma unroll
  for (int j = 0; j < 8; ++j) { m[j] = -3.0e38f; l[j] = 0.0f; }

  auto process = [&](v8f sv, int kbase) {
#pragma unroll
    for (int j = 0; j < 8; ++j) {
      int q = qbase + j + (hi << 3);
      int k = kbase + (lane & 15);
      float xs = sv[j] * 0.03125f;  // HEAD_SIZE^-0.5
      if (k > q) xs = -1.0e30f;
      float rm = xs;
#pragma unroll
      for (int off = 1; off < 16; off <<= 1)
        rm = fmaxf(rm, __shfl_xor(rm, off, 32));
      float mn    = fmaxf(m[j], rm);
      float alpha = __expf(m[j] - mn);
      float p     = __expf(xs - mn);
      float rs    = p;
#pragma unroll
      for (int off = 1; off < 16; off <<= 1)
        rs += __shfl_xor(rs, off, 32);
      l[j] = l[j] * alpha + rs;
      m[j] = mn;
      Sb[((size_t)b * T_ + q) * T_ + kbase + (lane & 15)] = (__bf16)xs;
    }
  };

  for (int kt = 0; kt <= qtile; kt += 2) {
    const int kbase = kt * 16;
    const bool has2 = (kt + 1) <= qtile;
    if (kt + 2 <= qtile)
      __builtin_prefetch(Kb + ((size_t)b * T_ + kbase + 32 + (lane & 15)) * H_, 0, 1);

    v8f s0 = {}, s1 = {};
    const __bf16* krow0 = Kb + ((size_t)b * T_ + kbase + (lane & 15)) * H_;
    const __bf16* krow1 = krow0 + (size_t)16 * H_;
    for (int h = 0; h < H_; h += 32) {
      V16U av;
      const char* ab = smem + ((size_t)(qslot + am) * H_ + h + a_k0) * 2;
      av.f4[0] = *(const f32x4*)(ab);
      av.f4[1] = *(const f32x4*)(ab + 32);
      V16U bv0, bv1;
      const f32x4* bp0 = (const f32x4*)(krow0 + h + b_k0);
      const f32x4* bp1 = (const f32x4*)(krow1 + h + b_k0);
      bv0.f4[0] = bp0[0]; bv0.f4[1] = bp0[2];
      bv1.f4[0] = bp1[0]; bv1.f4[1] = bp1[2];
      s0 = wmma_bf16(av.v, bv0.v, s0);
      s1 = wmma_bf16(av.v, bv1.v, s1);
    }
    process(s0, kbase);
    if (has2) process(s1, kbase + 16);
  }

  if ((lane & 15) == 0) {
#pragma unroll
    for (int j = 0; j < 8; ++j) {
      int q = qbase + j + (hi << 3);
      mbuf[b * T_ + q] = m[j];
      lbuf[b * T_ + q] = l[j];
    }
  }
}

// ---------------------------------------------------------------------------
// Kernel 3: O = softmax(S) * V.  exp fused into the A-fragment build; each
// wave computes a 16x128 slab (8 accumulators) to amortize the exp work 8x.
// ---------------------------------------------------------------------------
__global__ void __launch_bounds__(128)
pv_kernel(const __bf16* __restrict__ Sb, const __bf16* __restrict__ VTb,
          const float* __restrict__ mbuf, const float* __restrict__ lbuf,
          float* __restrict__ out) {
  const int lane = threadIdx.x & 31;
  const int hi   = lane >> 4;
  const int wid  = blockIdx.x * (blockDim.x >> 5) + (threadIdx.x >> 5);

  const int hblk = wid & 7;           // 8 slabs of 128 columns
  const int qt   = (wid >> 3) & 127;  // query tile
  const int b    = wid >> 10;

  const int qbase = qt * 16;
  const int hb    = hblk * 128;
  const int q     = qbase + (lane & 15);  // this lane's A row
  const float mrow = mbuf[b * T_ + q];
  const float invl = 1.0f / lbuf[b * T_ + q];

  v8f acc[8] = {{}, {}, {}, {}, {}, {}, {}, {}};
  const int nch = (qbase + 16 + 31) >> 5;  // causal key-chunk count (32 keys)

  const __bf16* srow = Sb + ((size_t)b * T_ + q) * T_;
  for (int c = 0; c < nch; ++c) {
    const int kchunk = c * 32;
    const int k0     = kchunk + (hi ? 8 : 0);

    // A fragment: p = exp(s - m) / l, masked causally
    V16U raw;
    raw.f4[0] = *(const f32x4*)(srow + k0);
    raw.f4[1] = *(const f32x4*)(srow + k0 + 16);
    v16bf av;
#pragma unroll
    for (int e = 0; e < 16; ++e) {
      int k = k0 + ((e < 8) ? e : (e + 8));
      float sv = (float)raw.v[e];
      float p  = (k <= q) ? __expf(sv - mrow) * invl : 0.0f;
      av[e] = (__bf16)p;
    }

    const int bko = kchunk + (hi ? 8 : 0);
#pragma unroll
    for (int n = 0; n < 8; ++n) {
      int h = hb + n * 16 + (lane & 15);
      const f32x4* vp = (const f32x4*)(VTb + ((size_t)b * H_ + h) * T_ + bko);
      V16U bv;
      bv.f4[0] = vp[0];
      bv.f4[1] = vp[2];
      acc[n] = wmma_bf16(av, bv.v, acc[n]);
    }
  }

#pragma unroll
  for (int n = 0; n < 8; ++n)
#pragma unroll
    for (int j = 0; j < 8; ++j) {
      int row = qbase + j + (hi << 3);
      out[((size_t)b * T_ + row) * H_ + hb + n * 16 + (lane & 15)] = acc[n][j];
    }
}

// ---------------------------------------------------------------------------
extern "C" void kernel_launch(void* const* d_in, const int* in_sizes, int n_in,
                              void* d_out, int out_size, void* d_ws, size_t ws_size,
                              hipStream_t stream) {
  const float* x  = (const float*)d_in[0];
  const float* Wk = (const float*)d_in[1];
  const float* Wv = (const float*)d_in[2];
  float* out = (float*)d_out;

  char* ws = (char*)d_ws;
  __bf16* Kb   = (__bf16*)(ws);                                   // 16 MiB
  __bf16* VTb  = (__bf16*)(ws + (size_t)16 * 1024 * 1024);        // 16 MiB
  __bf16* Sb   = (__bf16*)(ws + (size_t)32 * 1024 * 1024);        // 32 MiB
  float*  mbuf = (float*)(ws + (size_t)64 * 1024 * 1024);         // 32 KiB
  float*  lbuf = (float*)(ws + (size_t)64 * 1024 * 1024 + 65536); // 32 KiB

  // 2 matrices x 512 M-tiles x 16 N-groups = 16384 wave slabs / 4 waves
  proj_kv_kernel<<<4096, 128, 0, stream>>>(x, Wk, Wv, Kb, VTb);
  // B * (T/64) = 128 blocks, 128 KiB dynamic LDS (per-wave TDM Q staging)
  scores_kernel<<<128, 128, 128 * 1024, stream>>>(Kb, Sb, mbuf, lbuf);
  // B * (T/16) * (H/128) = 4096 wave slabs / 4 waves
  pv_kernel<<<1024, 128, 0, stream>>>(Sb, VTb, mbuf, lbuf, out);
}
